// MyNetwork_45621142618312
// MI455X (gfx1250) — matrix operand order repaired and checked by reading, not compiled, not run on hardware
//
#include <hip/hip_runtime.h>

// ---------------- problem constants ----------------
#define NN   50000
#define EE   800000
#define TT   5
#define FF   10
#define FOUT 2
#define GG   64
#define NL   2
#define C50  50      // T*F message channels
#define C64  64      // padded to 4 WMMA N-tiles
#define K32  32      // 3F=30 padded to one WMMA K step
#define NGRP (EE/16) // 50000 groups of 16 edges

typedef __attribute__((ext_vector_type(16))) _Float16 v16h;
typedef __attribute__((ext_vector_type(8)))  float    v8f;

// order-preserving float<->uint encoding for atomic min/max
__device__ __forceinline__ unsigned fenc(float x) {
  unsigned u = __float_as_uint(x);
  return (u & 0x80000000u) ? ~u : (u | 0x80000000u);
}
__device__ __forceinline__ float fdec(unsigned u) {
  return (u & 0x80000000u) ? __uint_as_float(u ^ 0x80000000u) : __uint_as_float(~u);
}

// ---------------- utility: fill ----------------
__global__ void fill_u32(unsigned* p, unsigned v, int n) {
  int i = blockIdx.x * blockDim.x + threadIdx.x;
  if (i < n) p[i] = v;
}

// ---------------- mlp1: 10 -> 5 (relu) -> 10 ----------------
__global__ void mlp1_kernel(const float* __restrict__ x,
                            const float* __restrict__ w1, const float* __restrict__ b1,
                            const float* __restrict__ w2, const float* __restrict__ b2,
                            float* __restrict__ h) {
  int n = blockIdx.x * blockDim.x + threadIdx.x;
  if (n >= NN) return;
  float xi[FF];
#pragma unroll
  for (int f = 0; f < FF; ++f) xi[f] = x[n * FF + f];
  float hid[5];
#pragma unroll
  for (int k = 0; k < 5; ++k) {
    float a = b1[k];
#pragma unroll
    for (int f = 0; f < FF; ++f) a += xi[f] * w1[f * 5 + k];
    hid[k] = fmaxf(a, 0.f);
  }
#pragma unroll
  for (int j = 0; j < FF; ++j) {
    float a = b2[j];
#pragma unroll
    for (int k = 0; k < 5; ++k) a += hid[k] * w2[k * FF + j];
    h[n * FF + j] = a;
  }
}

// ---------------- in-degree at dst ----------------
__global__ void deg_kernel(const int* __restrict__ ei, float* __restrict__ deg) {
  int e = blockIdx.x * blockDim.x + threadIdx.x;
  if (e >= EE) return;
  atomicAdd(&deg[ei[EE + e]], 1.0f);
}

// ---------------- sum of log(deg+1) over nodes ----------------
__global__ void avglog_kernel(const float* __restrict__ deg, float* __restrict__ out) {
  __shared__ float red[256];
  int n = blockIdx.x * blockDim.x + threadIdx.x;
  red[threadIdx.x] = (n < NN) ? __logf(deg[n] + 1.0f) : 0.f;
  __syncthreads();
  for (int s = 128; s > 0; s >>= 1) {
    if (threadIdx.x < s) red[threadIdx.x] += red[threadIdx.x + s];
    __syncthreads();
  }
  if (threadIdx.x == 0) atomicAdd(out, red[0]);
}

// ---------------- per-layer prep: padded combined weights, bias, edge table ----
// W50[k][c] = pre_w[l][c/10][k][c%10], padded to [32][64] with zeros.
// etab[a][j] = (edge_emb[a] @ enc_w[l] + enc_b[l])[j]  (only 4 distinct edge types)
__global__ void prep_kernel(int l,
                            const float* __restrict__ edge_emb,
                            const float* __restrict__ enc_w, const float* __restrict__ enc_b,
                            const float* __restrict__ pre_w, const float* __restrict__ pre_b,
                            float* __restrict__ W50, float* __restrict__ bias50,
                            float* __restrict__ etab,
                            float* __restrict__ bnsum, float* __restrict__ bnsq) {
  int t = threadIdx.x; // 64 threads
  for (int k = 0; k < K32; ++k) {
    float w = 0.f;
    if (t < C50 && k < 3 * FF) {
      int tw = t / FF, o = t % FF;
      w = pre_w[((l * TT + tw) * (3 * FF) + k) * FF + o];
    }
    W50[k * C64 + t] = w;
  }
  bias50[t] = (t < C50) ? pre_b[(l * TT + t / FF) * FF + (t % FF)] : 0.f;
  if (t < 4 * FF) {
    int a = t / FF, j = t % FF;
    float s = enc_b[l * FF + j];
    for (int f = 0; f < FF; ++f)
      s += edge_emb[a * FF + f] * enc_w[(l * FF + f) * FF + j];
    etab[t] = s;
  }
  if (t < FF) { bnsum[t] = 0.f; bnsq[t] = 0.f; }
}

// ---------------- fused edge message GEMM (WMMA) + scatter stats ----------------
// Each wave handles 16 edges/group: A[16x32 f16] x B[32x64 f16] via 4x
// v_wmma_f32_16x16x32_f16, then atomically folds D into sum/sq/min/max stats.
__global__ void __launch_bounds__(256)
edge_kernel(const float* __restrict__ h, const int* __restrict__ ei,
            const int* __restrict__ attr, const float* __restrict__ etab,
            const float* __restrict__ W50, const float* __restrict__ bias50,
            float* __restrict__ Ssum, float* __restrict__ Ssq,
            unsigned* __restrict__ Smin, unsigned* __restrict__ Smax) {
  __shared__ float msg[8][16 * K32];
  __shared__ int   dsts[8][16];

  const int lane = threadIdx.x & 31;
  const int wv   = threadIdx.x >> 5;
  const int gwave  = blockIdx.x * 8 + wv;   // 0..999
  const int nWaves = gridDim.x * 8;         // 1000 (uniform trip count: 50/wave)
  const int hh = lane >> 4;                 // half-wave select
  const int ln = lane & 15;

  // B fragments: 32x16 f16 per N-tile. Lane holds N=ln, K = 16*hh + 2v + {0,1}.
  v16h bf[4];
#pragma unroll
  for (int nt = 0; nt < 4; ++nt) {
#pragma unroll
    for (int v = 0; v < 8; ++v) {
      int k0 = hh * 16 + 2 * v;
      bf[nt][2 * v]     = (_Float16)W50[k0 * C64 + nt * 16 + ln];
      bf[nt][2 * v + 1] = (_Float16)W50[(k0 + 1) * C64 + nt * 16 + ln];
    }
  }
  float bias[4];
#pragma unroll
  for (int nt = 0; nt < 4; ++nt) bias[nt] = bias50[nt * 16 + ln];

  for (int grp = gwave; grp < NGRP; grp += nWaves) {
    // stage msg_in rows [16 edges x 32 feats] into LDS; each lane fills half a row
    const int e = grp * 16 + ln;
    const int sidx = ei[e];
    const int didx = ei[EE + e];
    const int a    = attr[e];
    if (hh == 0) dsts[wv][ln] = didx;
#pragma unroll
    for (int j = 0; j < 16; ++j) {
      int f = hh * 16 + j;
      float v;
      if (f < FF)            v = h[didx * FF + f];            // x_i (dst)
      else if (f < 2 * FF)   v = h[sidx * FF + (f - FF)];     // x_j (src)
      else if (f < 3 * FF)   v = etab[a * FF + (f - 2 * FF)]; // encoded edge attr
      else                   v = 0.f;                         // K padding
      msg[wv][ln * K32 + f] = v;
    }
    __syncthreads();

    // A fragment: 16x32 f16. Lane holds row M=ln, K = 16*hh + 2v + {0,1}.
    v16h af;
#pragma unroll
    for (int v = 0; v < 8; ++v) {
      int kb = hh * 16 + 2 * v;
      af[2 * v]     = (_Float16)msg[wv][ln * K32 + kb];
      af[2 * v + 1] = (_Float16)msg[wv][ln * K32 + kb + 1];
    }

    v8f acc[4];
#pragma unroll
    for (int nt = 0; nt < 4; ++nt) { v8f z = {}; acc[nt] = z; }
#pragma unroll
    for (int nt = 0; nt < 4; ++nt)
      acc[nt] = __builtin_amdgcn_wmma_f32_16x16x32_f16(
          /*neg_a=*/false, af, /*neg_b=*/false, bf[nt],
          /*c_mod=*/(short)0, acc[nt], /*reuse_a=*/false, /*reuse_b=*/false);

    // D layout: lane -> column c = nt*16 + ln; VGPR r -> edge row m = r + 8*hh
#pragma unroll
    for (int nt = 0; nt < 4; ++nt) {
      int c = nt * 16 + ln;
      if (c < C50) {
#pragma unroll
        for (int r = 0; r < 8; ++r) {
          int m = r + 8 * hh;
          int d = dsts[wv][m];
          float val = acc[nt][r] + bias[nt];
          int idx = d * C50 + c;
          atomicAdd(&Ssum[idx], val);
          atomicAdd(&Ssq[idx], val * val);
          unsigned encv = fenc(val);
          atomicMin(&Smin[idx], encv);
          atomicMax(&Smax[idx], encv);
        }
      }
    }
    __syncthreads();
  }
}

// ---------------- per-node: PNA scalers + post_nn + lin + BN-stat accumulation ----
__global__ void node_post_kernel(int l,
                                 const float* __restrict__ h, const float* __restrict__ deg,
                                 const float* __restrict__ avglog_sum,
                                 const float* __restrict__ Ssum, const float* __restrict__ Ssq,
                                 const unsigned* __restrict__ Smin, const unsigned* __restrict__ Smax,
                                 const float* __restrict__ post_w, const float* __restrict__ post_b,
                                 const float* __restrict__ lin_w, const float* __restrict__ lin_b,
                                 float* __restrict__ O,
                                 float* __restrict__ bnsum, float* __restrict__ bnsq) {
  int n = blockIdx.x * blockDim.x + threadIdx.x;
  if (n >= NN) return;
  float dg   = deg[n];
  float dcl  = fmaxf(dg, 1.0f);
  float invd = 1.0f / dcl;
  float avgl = avglog_sum[0] / (float)NN;
  float logd = __logf(dcl + 1.0f);
  float amp  = logd / avgl;
  float att  = avgl / logd;
  bool  has_in = dg > 0.f;

  float hn[FF];
#pragma unroll
  for (int f = 0; f < FF; ++f) hn[f] = h[n * FF + f];

  float flat[TT * FOUT];
#pragma unroll
  for (int t = 0; t < TT; ++t) {
    float agg[4][FF]; // mean, min, max, std
#pragma unroll
    for (int f = 0; f < FF; ++f) {
      int idx = n * C50 + t * FF + f;
      float mean  = Ssum[idx] * invd;
      float mean2 = Ssq[idx] * invd;
      float var   = fmaxf(mean2 - mean * mean, 0.f);
      agg[0][f] = mean;
      agg[1][f] = has_in ? fdec(Smin[idx]) : 0.f;
      agg[2][f] = has_in ? fdec(Smax[idx]) : 0.f;
      agg[3][f] = __fsqrt_rn(var + 1e-5f);
    }
    const float* pw = post_w + (size_t)(l * TT + t) * (13 * FF) * FOUT;
#pragma unroll
    for (int o = 0; o < FOUT; ++o) {
      float acc = post_b[(l * TT + t) * FOUT + o];
#pragma unroll
      for (int j = 0; j < FF; ++j) acc += hn[j] * pw[j * FOUT + o];
#pragma unroll
      for (int k = 0; k < 4 * FF; ++k) {
        float a = agg[k / FF][k % FF];
        acc += a * (pw[(FF + k) * FOUT + o] +
                    amp * pw[(5 * FF + k) * FOUT + o] +
                    att * pw[(9 * FF + k) * FOUT + o]);
      }
      flat[t * FOUT + o] = acc;
    }
  }
#pragma unroll
  for (int j = 0; j < FF; ++j) {
    float acc = lin_b[l * FF + j];
#pragma unroll
    for (int i = 0; i < TT * FOUT; ++i) acc += flat[i] * lin_w[(l * FF + i) * FF + j];
    O[n * FF + j] = acc;
    atomicAdd(&bnsum[j], acc);
    atomicAdd(&bnsq[j], acc * acc);
  }
}

// ---------------- apply train-mode BN + relu ----------------
__global__ void bn_relu_kernel(int l, const float* __restrict__ O,
                               const float* __restrict__ bnsum, const float* __restrict__ bnsq,
                               const float* __restrict__ gamma, const float* __restrict__ beta,
                               float* __restrict__ h) {
  int n = blockIdx.x * blockDim.x + threadIdx.x;
  if (n >= NN) return;
#pragma unroll
  for (int j = 0; j < FF; ++j) {
    float mu  = bnsum[j] / (float)NN;
    float var = bnsq[j] / (float)NN - mu * mu;
    float r   = __frsqrt_rn(var + 1e-5f);
    float v   = (O[n * FF + j] - mu) * r * gamma[l * FF + j] + beta[l * FF + j];
    h[n * FF + j] = fmaxf(v, 0.f);
  }
}

// ---------------- global add pool ----------------
__global__ void pool_kernel(const float* __restrict__ h, const int* __restrict__ batch,
                            float* __restrict__ g) {
  int n = blockIdx.x * blockDim.x + threadIdx.x;
  if (n >= NN) return;
  int b = batch[n];
#pragma unroll
  for (int f = 0; f < FF; ++f) atomicAdd(&g[b * FF + f], h[n * FF + f]);
}

// ---------------- head mlp2: 10 -> 5 (relu) -> 1 ----------------
__global__ void head_kernel(const float* __restrict__ g,
                            const float* __restrict__ w1, const float* __restrict__ b1,
                            const float* __restrict__ w2, const float* __restrict__ b2,
                            float* __restrict__ out) {
  int i = threadIdx.x;
  if (i >= GG) return;
  float hid[5];
#pragma unroll
  for (int k = 0; k < 5; ++k) {
    float a = b1[k];
#pragma unroll
    for (int f = 0; f < FF; ++f) a += g[i * FF + f] * w1[f * 5 + k];
    hid[k] = fmaxf(a, 0.f);
  }
  float o = b2[0];
#pragma unroll
  for (int k = 0; k < 5; ++k) o += hid[k] * w2[k];
  out[i] = o;
}

// ---------------- driver ----------------
extern "C" void kernel_launch(void* const* d_in, const int* in_sizes, int n_in,
                              void* d_out, int out_size, void* d_ws, size_t ws_size,
                              hipStream_t stream) {
  const float* x        = (const float*)d_in[0];
  const float* edge_emb = (const float*)d_in[1];
  const float* mlp1_w1  = (const float*)d_in[2];
  const float* mlp1_b1  = (const float*)d_in[3];
  const float* mlp1_w2  = (const float*)d_in[4];
  const float* mlp1_b2  = (const float*)d_in[5];
  const float* enc_w    = (const float*)d_in[6];
  const float* enc_b    = (const float*)d_in[7];
  const float* pre_w    = (const float*)d_in[8];
  const float* pre_b    = (const float*)d_in[9];
  const float* post_w   = (const float*)d_in[10];
  const float* post_b   = (const float*)d_in[11];
  const float* lin_w    = (const float*)d_in[12];
  const float* lin_b    = (const float*)d_in[13];
  const float* bn_gamma = (const float*)d_in[14];
  const float* bn_beta  = (const float*)d_in[15];
  const float* mlp2_w1  = (const float*)d_in[16];
  const float* mlp2_b1  = (const float*)d_in[17];
  const float* mlp2_w2  = (const float*)d_in[18];
  const float* mlp2_b2  = (const float*)d_in[19];
  const int* edge_index = (const int*)d_in[20];
  const int* edge_attr  = (const int*)d_in[21];
  const int* batch      = (const int*)d_in[22];
  float* out = (float*)d_out;

  float* ws = (float*)d_ws;
  size_t off = 0;
  float* H    = ws + off; off += (size_t)NN * FF;
  float* O    = ws + off; off += (size_t)NN * FF;
  float* DEG  = ws + off; off += NN;
  float* SSUM = ws + off; off += (size_t)NN * C50;
  float* SSQ  = ws + off; off += (size_t)NN * C50;
  unsigned* SMIN = (unsigned*)(ws + off); off += (size_t)NN * C50;
  unsigned* SMAX = (unsigned*)(ws + off); off += (size_t)NN * C50;
  float* W50  = ws + off; off += K32 * C64;
  float* B50  = ws + off; off += C64;
  float* ETAB = ws + off; off += 4 * FF;
  float* BNS  = ws + off; off += FF;
  float* BNQ  = ws + off; off += FF;
  float* ALOG = ws + off; off += 1;
  float* GP   = ws + off; off += GG * FF;

  const int TPB = 256;
  // fresh accumulators every call (graph-replay safe)
  fill_u32<<<(NN + TPB - 1) / TPB, TPB, 0, stream>>>((unsigned*)DEG, 0u, NN);
  fill_u32<<<1, TPB, 0, stream>>>((unsigned*)ALOG, 0u, 1);
  fill_u32<<<1, TPB, 0, stream>>>((unsigned*)GP, 0u, GG * FF);

  mlp1_kernel<<<(NN + TPB - 1) / TPB, TPB, 0, stream>>>(x, mlp1_w1, mlp1_b1, mlp1_w2, mlp1_b2, H);
  deg_kernel<<<(EE + TPB - 1) / TPB, TPB, 0, stream>>>(edge_index, DEG);
  avglog_kernel<<<(NN + TPB - 1) / TPB, TPB, 0, stream>>>(DEG, ALOG);

  for (int l = 0; l < NL; ++l) {
    const int M = NN * C50;
    fill_u32<<<(M + TPB - 1) / TPB, TPB, 0, stream>>>((unsigned*)SSUM, 0u, M);
    fill_u32<<<(M + TPB - 1) / TPB, TPB, 0, stream>>>((unsigned*)SSQ, 0u, M);
    fill_u32<<<(M + TPB - 1) / TPB, TPB, 0, stream>>>(SMIN, 0xFFFFFFFFu, M);
    fill_u32<<<(M + TPB - 1) / TPB, TPB, 0, stream>>>(SMAX, 0u, M);
    prep_kernel<<<1, 64, 0, stream>>>(l, edge_emb, enc_w, enc_b, pre_w, pre_b,
                                      W50, B50, ETAB, BNS, BNQ);
    edge_kernel<<<125, 256, 0, stream>>>(H, edge_index, edge_attr, ETAB, W50, B50,
                                         SSUM, SSQ, SMIN, SMAX);
    node_post_kernel<<<(NN + TPB - 1) / TPB, TPB, 0, stream>>>(
        l, H, DEG, ALOG, SSUM, SSQ, SMIN, SMAX,
        post_w, post_b, lin_w, lin_b, O, BNS, BNQ);
    bn_relu_kernel<<<(NN + TPB - 1) / TPB, TPB, 0, stream>>>(l, O, BNS, BNQ,
                                                             bn_gamma, bn_beta, H);
  }
  pool_kernel<<<(NN + TPB - 1) / TPB, TPB, 0, stream>>>(H, batch, GP);
  head_kernel<<<1, 64, 0, stream>>>(GP, mlp2_w1, mlp2_b1, mlp2_w2, mlp2_b2, out);
}
